// CausalSelfAttention_87720412054208
// MI455X (gfx1250) — matrix operand order repaired and checked by reading, not compiled, not run on hardware
//
#include <hip/hip_runtime.h>
#include <hip/hip_bf16.h>

typedef __attribute__((ext_vector_type(16))) __bf16 v16bf;
typedef __attribute__((ext_vector_type(8)))  float  v8f;

union FragAB { v16bf v; uint4 q[2]; };           // 32B: one WMMA bf16 A/B operand

#define WMMA_BF16(A, B, C) \
  __builtin_amdgcn_wmma_f32_16x16x32_bf16(false, (A), false, (B), (short)0, (C), false, false)

#define D_MODEL  1024
#define N_HEADS  16
#define HEAD_DIM 64
#define BATCH    2
#define SEQ      2048
#define ROWS     (BATCH * SEQ)   // 4096

// ------------------------------------------------------------ CDNA5 async Global->LDS
// Copies 16 bytes per lane straight into LDS (ASYNCcnt), no VGPR round trip.
__device__ __forceinline__ void async_copy16(void* lds, const void* g) {
  unsigned            l = (unsigned)(unsigned long long)lds;       // addr[31:0] == LDS offset
  unsigned long long  a = (unsigned long long)g;
  asm volatile("global_load_async_to_lds_b128 %0, %1, off" :: "v"(l), "v"(a) : "memory");
}
#define ASYNC_WAIT_4() asm volatile("s_wait_asynccnt 4" ::: "memory")
#define ASYNC_WAIT_0() asm volatile("s_wait_asynccnt 0" ::: "memory")

// ------------------------------------------------------------ 16-lane butterfly via DPP16
template<int CTRL>
__device__ __forceinline__ float dpp_mov(float x) {
  int r = __builtin_amdgcn_update_dpp(0, __builtin_bit_cast(int, x), CTRL, 0xf, 0xf, true);
  return __builtin_bit_cast(float, r);
}
__device__ __forceinline__ float red16_max(float x) {
  x = fmaxf(x, dpp_mov<0xB1>(x));    // quad_perm [1,0,3,2]  (xor 1)
  x = fmaxf(x, dpp_mov<0x4E>(x));    // quad_perm [2,3,0,1]  (xor 2)
  x = fmaxf(x, dpp_mov<0x141>(x));   // row_half_mirror      (xor 4 effective)
  x = fmaxf(x, dpp_mov<0x140>(x));   // row_mirror           (xor 8 effective)
  return x;
}
__device__ __forceinline__ float red16_sum(float x) {
  x += dpp_mov<0xB1>(x);
  x += dpp_mov<0x4E>(x);
  x += dpp_mov<0x141>(x);
  x += dpp_mov<0x140>(x);
  return x;
}

// ------------------------------------------------------------ conversions
__global__ void cvt_f32_bf16(const float* __restrict__ in, __bf16* __restrict__ out, int n) {
  int i = blockIdx.x * blockDim.x + threadIdx.x;
  if (i < n) out[i] = (__bf16)in[i];
}
// out[N][K] = in[K][N]^T   (coalesced writes; one-time cost)
__global__ void cvt_f32_bf16_T(const float* __restrict__ in, __bf16* __restrict__ out,
                               int K, int N) {
  int i = blockIdx.x * blockDim.x + threadIdx.x;
  if (i < K * N) {
    int n = i >> 10, k = i & 1023;   // K == 1024 always here
    out[(size_t)n * K + k] = (__bf16)in[(size_t)k * N + n];
  }
}

// ------------------------------------------------------------ bf16 GEMM (async double-buffered)
// C[M,N] = A[M,K] * BT[N,K]^T. Block 128x128, K-step 32, 256 threads (8 waves, 4x2).
// MODE 0: scatter qkv; q,k -> [B,H,T,Dh]; v -> [B,H,Dh,T] (pre-transposed for attention).
// MODE 1: fp32 out [M,N].
template<int MODE>
__global__ __launch_bounds__(256) void gemm_bf16(
    const __bf16* __restrict__ A, const __bf16* __restrict__ BT,
    int K, int N,
    __bf16* __restrict__ qp, __bf16* __restrict__ kp, __bf16* __restrict__ vp,
    float* __restrict__ outp)
{
  __shared__ __attribute__((aligned(16))) __bf16 aS[2][128][40];  // [m][k]
  __shared__ __attribute__((aligned(16))) __bf16 bS[2][128][40];  // [n][k]

  const int tid  = threadIdx.x;
  const int lane = tid & 31;
  const int wid  = tid >> 5;
  const int wm   = wid & 3;
  const int wn   = wid >> 2;
  const int m0   = blockIdx.y * 128;
  const int n0   = blockIdx.x * 128;

  const int srow = tid >> 1;
  const int soff = (tid & 1) * 16;

  auto stage = [&](int buf, int kt) {        // 4 async b128 copies per thread
    const __bf16* ga = A  + (size_t)(m0 + srow) * K + kt * 32 + soff;
    async_copy16(&aS[buf][srow][soff],     ga);
    async_copy16(&aS[buf][srow][soff + 8], ga + 8);
    const __bf16* gb = BT + (size_t)(n0 + srow) * K + kt * 32 + soff;
    async_copy16(&bS[buf][srow][soff],     gb);
    async_copy16(&bS[buf][srow][soff + 8], gb + 8);
  };

  v8f c[2][4];
  #pragma unroll
  for (int i = 0; i < 2; ++i)
    #pragma unroll
    for (int j = 0; j < 4; ++j)
      c[i][j] = (v8f){0.f,0.f,0.f,0.f,0.f,0.f,0.f,0.f};

  const int nk = K / 32;
  stage(0, 0);
  for (int kt = 0; kt < nk; ++kt) {
    const int buf = kt & 1;
    if (kt + 1 < nk) { stage(buf ^ 1, kt + 1); ASYNC_WAIT_4(); }
    else             { ASYNC_WAIT_0(); }
    __syncthreads();

    FragAB a[2], b[4];
    const int koffA = (lane < 16) ? 0 : 8;   // A layout: K {0..7,16..23} / {8..15,24..31}
    #pragma unroll
    for (int i = 0; i < 2; ++i) {
      int row = wm * 32 + i * 16 + (lane & 15);
      a[i].q[0] = *(const uint4*)&aS[buf][row][koffA];
      a[i].q[1] = *(const uint4*)&aS[buf][row][koffA + 16];
    }
    const int koffB = (lane < 16) ? 0 : 8;   // B layout: K {0..7,16..23} / {8..15,24..31}
    #pragma unroll
    for (int j = 0; j < 4; ++j) {
      int n = wn * 64 + j * 16 + (lane & 15);
      b[j].q[0] = *(const uint4*)&bS[buf][n][koffB];
      b[j].q[1] = *(const uint4*)&bS[buf][n][koffB + 16];
    }
    #pragma unroll
    for (int i = 0; i < 2; ++i)
      #pragma unroll
      for (int j = 0; j < 4; ++j)
        c[i][j] = WMMA_BF16(a[i].v, b[j].v, c[i][j]);
    __syncthreads();
  }

  #pragma unroll
  for (int i = 0; i < 2; ++i)
    #pragma unroll
    for (int j = 0; j < 4; ++j)
      #pragma unroll
      for (int r = 0; r < 8; ++r) {
        int m = m0 + wm * 32 + i * 16 + r + ((lane >= 16) ? 8 : 0);
        int n = n0 + wn * 64 + j * 16 + (lane & 15);
        float val = c[i][j][r];
        if (MODE == 0) {
          int part = n >> 10, cc = n & 1023;
          int head = cc >> 6, dd = cc & 63;
          int bb = m >> 11,  tt = m & 2047;
          int bh = bb * N_HEADS + head;
          if (part == 0)      qp[((size_t)bh * SEQ + tt) * HEAD_DIM + dd] = (__bf16)(val * 0.125f);
          else if (part == 1) kp[((size_t)bh * SEQ + tt) * HEAD_DIM + dd] = (__bf16)val;
          else                vp[((size_t)bh * HEAD_DIM + dd) * SEQ + tt] = (__bf16)val; // V^T
        } else {
          outp[(size_t)m * N + n] = val;
        }
      }
}

// ------------------------------------------------------------ flash attention
// Grid (T/64, B*H); 128 threads = 4 waves; 32-key tiles, async double-buffered K/V,
// online softmax with DPP16 reductions. V arrives pre-transposed [B,H,Dh,T].
__global__ __launch_bounds__(128) void attn_kernel(
    const __bf16* __restrict__ Q, const __bf16* __restrict__ Km,
    const __bf16* __restrict__ Vt, __bf16* __restrict__ Y)
{
  __shared__ __attribute__((aligned(16))) __bf16 kS[2][32][72];    // [tk][d]
  __shared__ __attribute__((aligned(16))) __bf16 vS[2][64][40];    // [d][tk]
  __shared__ __attribute__((aligned(16))) __bf16 pS[4][16][40];    // per-wave P 16x32

  const int tid  = threadIdx.x;
  const int lane = tid & 31;
  const int wid  = tid >> 5;
  const int qb   = blockIdx.x;
  const int bh   = blockIdx.y;
  const int b    = bh >> 4;
  const int h    = bh & 15;
  const int qrow0 = qb * 64 + wid * 16;
  const float NEG_INF = -__builtin_inff();

  const int krow = tid >> 2, kcol = (tid & 3) * 16;   // K tile 32x64
  const int vrow = tid >> 1, vcol = (tid & 1) * 16;   // V tile 64x32

  auto stage = [&](int buf, int j) {                  // 4 async b128 copies per thread
    const __bf16* gk = Km + ((size_t)bh * SEQ + j + krow) * HEAD_DIM + kcol;
    async_copy16(&kS[buf][krow][kcol],     gk);
    async_copy16(&kS[buf][krow][kcol + 8], gk + 8);
    const __bf16* gv = Vt + ((size_t)bh * HEAD_DIM + vrow) * SEQ + j + vcol;
    async_copy16(&vS[buf][vrow][vcol],     gv);
    async_copy16(&vS[buf][vrow][vcol + 8], gv + 8);
  };

  // Q fragments in registers for the whole loop (scale folded in at QKV store)
  FragAB qa[2];
  {
    const __bf16* qptr = Q + ((size_t)bh * SEQ + qrow0 + (lane & 15)) * HEAD_DIM;
    const int koff = (lane < 16) ? 0 : 8;
    #pragma unroll
    for (int kb = 0; kb < 2; ++kb) {
      qa[kb].q[0] = *(const uint4*)(qptr + kb * 32 + koff);
      qa[kb].q[1] = *(const uint4*)(qptr + kb * 32 + koff + 16);
    }
  }

  v8f o[4];
  #pragma unroll
  for (int jn = 0; jn < 4; ++jn) o[jn] = (v8f){0.f,0.f,0.f,0.f,0.f,0.f,0.f,0.f};
  float rmax[8], rsum[8];
  #pragma unroll
  for (int r = 0; r < 8; ++r) { rmax[r] = NEG_INF; rsum[r] = 0.f; }

  const int koffB = (lane < 16) ? 0 : 8;
  const int nj    = (qb + 1) * 2;          // number of 32-wide key tiles (causal bound)

  stage(0, 0);
  for (int jt = 0; jt < nj; ++jt) {
    const int j   = jt * 32;
    const int buf = jt & 1;
    if (jt + 1 < nj) { stage(buf ^ 1, (jt + 1) * 32); ASYNC_WAIT_4(); }
    else             { ASYNC_WAIT_0(); }
    __syncthreads();

    // S[16x32] = Q * K^T : 4 WMMAs
    v8f s0 = (v8f){0.f,0.f,0.f,0.f,0.f,0.f,0.f,0.f};
    v8f s1 = s0;
    #pragma unroll
    for (int kb = 0; kb < 2; ++kb) {
      FragAB bk0, bk1;
      int tkA = lane & 15, tkB = 16 + (lane & 15);
      bk0.q[0] = *(const uint4*)&kS[buf][tkA][kb * 32 + koffB];
      bk0.q[1] = *(const uint4*)&kS[buf][tkA][kb * 32 + koffB + 16];
      bk1.q[0] = *(const uint4*)&kS[buf][tkB][kb * 32 + koffB];
      bk1.q[1] = *(const uint4*)&kS[buf][tkB][kb * 32 + koffB + 16];
      s0 = WMMA_BF16(qa[kb].v, bk0.v, s0);
      s1 = WMMA_BF16(qa[kb].v, bk1.v, s1);
    }

    // online softmax: all cross-lane traffic via DPP16 (pure VALU)
    #pragma unroll
    for (int r = 0; r < 8; ++r) {
      int mloc = r + ((lane >= 16) ? 8 : 0);
      int qidx = qb * 64 + wid * 16 + mloc;
      int tk0  = j + (lane & 15);
      float x0 = (tk0      <= qidx) ? s0[r] : NEG_INF;
      float x1 = (tk0 + 16 <= qidx) ? s1[r] : NEG_INF;
      float loc = red16_max(fmaxf(x0, x1));
      float nm  = fmaxf(rmax[r], loc);
      float sc  = __expf(rmax[r] - nm);
      float p0  = __expf(x0 - nm);
      float p1  = __expf(x1 - nm);
      float ls  = red16_sum(p0 + p1);
      rsum[r] = rsum[r] * sc + ls;
      rmax[r] = nm;
      #pragma unroll
      for (int jn = 0; jn < 4; ++jn) o[jn][r] *= sc;
      pS[wid][mloc][lane & 15]        = (__bf16)p0;
      pS[wid][mloc][16 + (lane & 15)] = (__bf16)p1;
    }
    __syncthreads();   // orders pS write -> read (and keeps waves in lockstep)

    // O[16x64] += P[16x32] * V[32x64] : 4 WMMAs
    FragAB pa;
    const int koffA = (lane < 16) ? 0 : 8;
    pa.q[0] = *(const uint4*)&pS[wid][lane & 15][koffA];
    pa.q[1] = *(const uint4*)&pS[wid][lane & 15][koffA + 16];
    #pragma unroll
    for (int jn = 0; jn < 4; ++jn) {
      FragAB bv;
      int d = jn * 16 + (lane & 15);
      bv.q[0] = *(const uint4*)&vS[buf][d][koffB];
      bv.q[1] = *(const uint4*)&vS[buf][d][koffB + 16];
      o[jn] = WMMA_BF16(pa.v, bv.v, o[jn]);
    }
    __syncthreads();
  }

  #pragma unroll
  for (int jn = 0; jn < 4; ++jn)
    #pragma unroll
    for (int r = 0; r < 8; ++r) {
      int mloc = r + ((lane >= 16) ? 8 : 0);
      int t = qb * 64 + wid * 16 + mloc;
      int d = jn * 16 + (lane & 15);
      float val = o[jn][r] / rsum[r];
      Y[((size_t)(b * SEQ + t)) * D_MODEL + h * HEAD_DIM + d] = (__bf16)val;
    }
}

// ------------------------------------------------------------ launch
extern "C" void kernel_launch(void* const* d_in, const int* in_sizes, int n_in,
                              void* d_out, int out_size, void* d_ws, size_t ws_size,
                              hipStream_t stream) {
  const float* x     = (const float*)d_in[0];
  const float* w_qkv = (const float*)d_in[1];
  const float* w_out = (const float*)d_in[2];
  float* out = (float*)d_out;

  char* w = (char*)d_ws;
  size_t off = 0;
  __bf16* xb     = (__bf16*)(w + off); off += (size_t)ROWS * D_MODEL * 2;        // 8 MB
  __bf16* wqkvT  = (__bf16*)(w + off); off += (size_t)3 * D_MODEL * D_MODEL * 2; // 6 MB  [3072][1024]
  __bf16* woutT  = (__bf16*)(w + off); off += (size_t)D_MODEL * D_MODEL * 2;     // 2 MB  [1024][1024]
  __bf16* qbuf   = (__bf16*)(w + off); off += (size_t)ROWS * D_MODEL * 2;        // 8 MB  [B,H,T,Dh]
  __bf16* kbuf   = (__bf16*)(w + off); off += (size_t)ROWS * D_MODEL * 2;        // 8 MB  [B,H,T,Dh]
  __bf16* vbuf   = (__bf16*)(w + off); off += (size_t)ROWS * D_MODEL * 2;        // 8 MB  [B,H,Dh,T]
  __bf16* ybuf   = (__bf16*)(w + off); off += (size_t)ROWS * D_MODEL * 2;        // 8 MB  [B,T,C]

  const int nx = ROWS * D_MODEL;
  cvt_f32_bf16  <<<(nx + 255) / 256, 256, 0, stream>>>(x, xb, nx);
  cvt_f32_bf16_T<<<(3 * D_MODEL * D_MODEL + 255) / 256, 256, 0, stream>>>(w_qkv, wqkvT, D_MODEL, 3 * D_MODEL);
  cvt_f32_bf16_T<<<(D_MODEL * D_MODEL + 255) / 256, 256, 0, stream>>>(w_out, woutT, D_MODEL, D_MODEL);

  gemm_bf16<0><<<dim3(3 * D_MODEL / 128, ROWS / 128), 256, 0, stream>>>(
      xb, wqkvT, D_MODEL, 3 * D_MODEL, qbuf, kbuf, vbuf, nullptr);

  attn_kernel<<<dim3(SEQ / 64, BATCH * N_HEADS), 128, 0, stream>>>(
      qbuf, kbuf, vbuf, ybuf);

  gemm_bf16<1><<<dim3(D_MODEL / 128, ROWS / 128), 256, 0, stream>>>(
      ybuf, woutT, D_MODEL, D_MODEL, nullptr, nullptr, nullptr, out);
}
